// SAM_R_60112362275681
// MI455X (gfx1250) — matrix operand order repaired and checked by reading, not compiled, not run on hardware
//
#include <hip/hip_runtime.h>
#include <hip/hip_bf16.h>
#include <math.h>

// ---------------------------------------------------------------------------
// Problem constants (from reference)
// ---------------------------------------------------------------------------
typedef __bf16 bf16;
typedef __bf16 v16bf __attribute__((ext_vector_type(16)));
typedef __bf16 v8bf  __attribute__((ext_vector_type(8)));
typedef float  v8f   __attribute__((ext_vector_type(8)));

constexpr int Bn    = 2;
constexpr int Cc    = 256;
constexpr int Hs    = 48;
constexpr int Wss   = 48;
constexpr int Ms    = Hs * Wss;      // 2304 tokens
constexpr int HEADS = 8;
constexpr int DH    = Cc / HEADS;    // 32 == WMMA bf16 K — one wmma per logits tile
constexpr int OT    = Cc / 16;       // 16 output-row tiles
constexpr int MT    = Ms / 16;       // 144 token tiles
constexpr int TILES = OT * MT;       // 2304 tiles per batch

// ---------------------------------------------------------------------------
// WMMA helpers (gfx1250 wave32 layouts, CDNA5 ISA)
// ---------------------------------------------------------------------------
__device__ inline v8f wmma_bf(v16bf a, v16bf b, v8f c) {
  return __builtin_amdgcn_wmma_f32_16x16x32_bf16(
      /*neg_a=*/false, a, /*neg_b=*/false, b,
      /*c_mod=*/(short)0, c, /*reuse_a=*/false, /*reuse_b=*/false);
}

__device__ inline v16bf cat8(v8bf lo, v8bf hi) {
  return __builtin_shufflevector(lo, hi, 0,1,2,3,4,5,6,7,8,9,10,11,12,13,14,15);
}

// A fragment 16(m) x 32(k), row-major source: element (m,k) = src[(row0+m)*ld + k0+k]
// Lane L: m = L&15 ; K groups {kb..kb+7} and {16+kb..16+kb+7}, kb = (L>>4)*8
// -> two contiguous 16B loads per lane.
__device__ inline v16bf load_A_rm(const bf16* src, int row0, int k0, int ld) {
  const int lane = threadIdx.x & 31;
  const bf16* p = src + (size_t)(row0 + (lane & 15)) * ld + k0 + ((lane >> 4) << 3);
  v8bf lo = *reinterpret_cast<const v8bf*>(p);
  v8bf hi = *reinterpret_cast<const v8bf*>(p + 16);
  return cat8(lo, hi);
}

// B fragment 32(k) x 16(n), n-major source (contiguous k):
// element (k,n) = src[(n0+n)*ld + k0+k].  Lane L: n = L&15 ; K groups {kb..kb+7} and {16+kb..16+kb+7}, kb = (L>>4)*8
// -> two contiguous 16B loads per lane.
__device__ inline v16bf load_B_nm(const bf16* src, int k0, int n0, int ld) {
  const int lane = threadIdx.x & 31;
  const bf16* p = src + (size_t)(n0 + (lane & 15)) * ld + k0 + ((lane >> 4) << 3);
  v8bf lo = *reinterpret_cast<const v8bf*>(p);
  v8bf hi = *reinterpret_cast<const v8bf*>(p + 16);
  return cat8(lo, hi);
}

// half-wave (16-lane) reductions: rows of a C/D tile live across 16 lanes
__device__ inline float red_max16(float v) {
  #pragma unroll
  for (int off = 8; off; off >>= 1) v = fmaxf(v, __shfl_xor(v, off, 32));
  return v;
}
__device__ inline float red_sum16(float v) {
  #pragma unroll
  for (int off = 8; off; off >>= 1) v += __shfl_xor(v, off, 32);
  return v;
}

// ---------------------------------------------------------------------------
// Kernel 1: convert the four 256x256 fp32 weight matrices to bf16
// ---------------------------------------------------------------------------
__global__ void cvtw_kernel(const float* __restrict__ w0, const float* __restrict__ w1,
                            const float* __restrict__ w2, const float* __restrict__ w3,
                            bf16* __restrict__ out) {
  int idx = blockIdx.x * 256 + threadIdx.x;      // 4*65536 total
  int sel = idx >> 16, i = idx & 65535;
  const float* w = (sel == 0) ? w0 : (sel == 1) ? w1 : (sel == 2) ? w2 : w3;
  out[idx] = (bf16)w[i];
}

// ---------------------------------------------------------------------------
// Kernel 2: fused inference-BN, fp32 -> bf16, token-major output xnT[b][m][c]
// ---------------------------------------------------------------------------
__global__ void bn_kernel(const float* __restrict__ x, const float* __restrict__ gamma,
                          const float* __restrict__ beta, const float* __restrict__ mean,
                          const float* __restrict__ var, bf16* __restrict__ xnT) {
  int idx = blockIdx.x * 256 + threadIdx.x;      // Bn*Ms*Cc
  int c = idx & 255;
  int t = idx >> 8;                              // b*Ms + m
  int m = t % Ms, b = t / Ms;
  float inv  = gamma[c] * rsqrtf(var[c] + 1e-5f);
  float bias = beta[c] - mean[c] * inv;
  float v = x[((size_t)b * Cc + c) * Ms + m];
  xnT[(size_t)t * Cc + c] = (bf16)(v * inv + bias);
}

// ---------------------------------------------------------------------------
// Kernel 3: L2-normalized 2D sinusoidal PE table, rl2[m][c] (sum r^2 == 128)
// ---------------------------------------------------------------------------
__global__ void pe_kernel(bf16* __restrict__ rl2) {
  int idx = blockIdx.x * 256 + threadIdx.x;      // Ms*Cc, layout m-major
  int c = idx & 255, m = idx >> 8;
  int xp = m % Wss, yp = m / Wss;
  int pos = (c < 128) ? xp : yp;
  int cc  = (c < 128) ? c : c - 128;
  int k   = cc >> 1;
  float dv = __expf((float)(k << 1) * (-9.210340371976184f / 128.0f)); // -ln(1e4)/d
  float tv = (float)pos * dv;
  float r  = (cc & 1) ? __cosf(tv) : __sinf(tv);
  rl2[idx] = (bf16)(r * 0.0883883459f);          // 1/sqrt(128 + 1e-6)
}

// ---------------------------------------------------------------------------
// Kernel 4: generic 256xKx M GEMM via WMMA: out[o][m] = sum_c W[o][c] * XT[m][c]
// Exactly one non-null output: Ytok (token-major bf16), Ycm (chan-major bf16),
// or Yf (chan-major fp32, final result).
// ---------------------------------------------------------------------------
__global__ void gemm_kernel(const bf16* __restrict__ Wb, const bf16* __restrict__ XT,
                            bf16* __restrict__ Ycm, bf16* __restrict__ Ytok,
                            float* __restrict__ Yf) {
  const int lane = threadIdx.x & 31;
  const int tile = blockIdx.x * 4 + (threadIdx.x >> 5);  // 4 waves / block
  const int ot = tile / MT, mt = tile % MT;
  const int b  = blockIdx.z;
  const bf16* Xb = XT + (size_t)b * Ms * Cc;

  v8f acc = {0.f, 0.f, 0.f, 0.f, 0.f, 0.f, 0.f, 0.f};
  #pragma unroll
  for (int k0 = 0; k0 < Cc; k0 += 32) {
    v16bf aa = load_A_rm(Wb, ot * 16, k0, Cc);           // W tile 16x32
    v16bf bb = load_B_nm(Xb, k0, mt * 16, Cc);           // x tile 32x16
    acc = wmma_bf(aa, bb, acc);
  }

  const int colm = mt * 16 + (lane & 15);                // token index
  const int row0 = ot * 16 + ((lane >> 4) << 3);         // out-channel base
  if (Yf) {
    float* y = Yf + (size_t)b * Cc * Ms;
    #pragma unroll
    for (int r = 0; r < 8; ++r) y[(size_t)(row0 + r) * Ms + colm] = acc[r];
  } else if (Ytok) {
    bf16* y = Ytok + (size_t)b * Ms * Cc;
    #pragma unroll
    for (int r = 0; r < 8; ++r) y[(size_t)colm * Cc + row0 + r] = (bf16)acc[r];
  } else {
    bf16* y = Ycm + (size_t)b * Cc * Ms;
    #pragma unroll
    for (int r = 0; r < 8; ++r) y[(size_t)(row0 + r) * Ms + colm] = (bf16)acc[r];
  }
}

// ---------------------------------------------------------------------------
// Kernel 5: per-token L2 norm of z over all 256 channels, add PE, -> zrT[b][m][c]
// ---------------------------------------------------------------------------
__global__ void norm_kernel(const bf16* __restrict__ zT, const bf16* __restrict__ rl2,
                            bf16* __restrict__ zrT) {
  int idx = blockIdx.x * 256 + threadIdx.x;      // Bn*Ms
  int m = idx % Ms;
  const bf16* z = zT + (size_t)idx * Cc;
  float s = 0.f;
  #pragma unroll 8
  for (int c = 0; c < Cc; ++c) { float v = (float)z[c]; s += v * v; }
  float rn = rsqrtf(s + 1e-6f);
  bf16* o = zrT + (size_t)idx * Cc;
  const bf16* r = rl2 + (size_t)m * Cc;
  #pragma unroll 8
  for (int c = 0; c < Cc; ++c) o[c] = (bf16)((float)z[c] * rn + (float)r[c]);
}

// ---------------------------------------------------------------------------
// Kernel 6: flash-style attention. One wave owns a 16-row (token) tile of one
// (batch, head). dh == 32 == WMMA K, so each 16x16 logits tile is ONE wmma.
// Online softmax in registers; P transposed C-layout -> A-layout through LDS.
// ---------------------------------------------------------------------------
__global__ void attn_kernel(const bf16* __restrict__ vT, const bf16* __restrict__ zrT,
                            const bf16* __restrict__ qcm, bf16* __restrict__ yT) {
  alignas(16) __shared__ bf16 plds[4][16 * 32];
  const int lane = threadIdx.x & 31;
  const int wid  = threadIdx.x >> 5;
  const int mt   = blockIdx.x * 4 + wid;         // 0..143
  const int head = blockIdx.y;
  const int b    = blockIdx.z;
  const int ch0  = head * DH;
  const int m0   = mt * 16;

  const bf16* vTb  = vT  + (size_t)b * Ms * Cc;
  const bf16* zrTb = zrT + (size_t)b * Ms * Cc;
  const bf16* qb   = qcm + (size_t)b * Cc * Ms;
  bf16*       yTb  = yT  + (size_t)b * Ms * Cc;
  bf16*       pl   = plds[wid];

  // vh^T tile: A 16(m) x 32(k=c'), token-major source -> contiguous loads
  v16bf a = load_A_rm(vTb, m0, ch0, Cc);

  v8f O0 = {0.f,0.f,0.f,0.f,0.f,0.f,0.f,0.f};
  v8f O1 = {0.f,0.f,0.f,0.f,0.f,0.f,0.f,0.f};
  const v8f zc = {0.f,0.f,0.f,0.f,0.f,0.f,0.f,0.f};
  float mr[8], lr[8];
  #pragma unroll
  for (int r = 0; r < 8; ++r) { mr[r] = -1e30f; lr[r] = 0.f; }
  const float sc = 0.17677669529663687f;         // 1/sqrt(dh)

  for (int n0 = 0; n0 < Ms; n0 += 32) {
    // logits: S = vh^T (16xdh) * zrh (dh x 32) -> two 16x16 tiles, one wmma each
    v16bf b0 = load_B_nm(zrTb, ch0, n0,      Cc);
    v16bf b1 = load_B_nm(zrTb, ch0, n0 + 16, Cc);
    v8f s0 = wmma_bf(a, b0, zc);
    v8f s1 = wmma_bf(a, b1, zc);

    // online softmax per row (rows of C/D tile span 16 lanes of one VGPR)
    #pragma unroll
    for (int r = 0; r < 8; ++r) {
      float x0 = s0[r] * sc, x1 = s1[r] * sc;
      float rm = red_max16(fmaxf(x0, x1));
      float mn = fmaxf(mr[r], rm);
      float al = __expf(mr[r] - mn);
      mr[r] = mn;
      float p0 = __expf(x0 - mn), p1 = __expf(x1 - mn);
      lr[r] = lr[r] * al + red_sum16(p0 + p1);
      O0[r] *= al; O1[r] *= al;
      // stash P row-major in this wave's LDS slice (C-layout -> A-layout)
      int mrow = ((lane >> 4) << 3) + r;
      int ncol = lane & 15;
      pl[mrow * 32 + ncol]      = (bf16)p0;
      pl[mrow * 32 + 16 + ncol] = (bf16)p1;
    }

    // O += P (16x32) * qh^T (32 x dh): A from LDS, B contiguous-K from q[c][m]
    v16bf p  = load_A_rm(pl, 0, 0, 32);
    v16bf q0 = load_B_nm(qb + (size_t)ch0 * Ms,        n0, 0, Ms);
    v16bf q1 = load_B_nm(qb + (size_t)(ch0 + 16) * Ms, n0, 0, Ms);
    O0 = wmma_bf(p, q0, O0);
    O1 = wmma_bf(p, q1, O1);
  }

  // finalize: divide by row sum, write yT[b][m][c]
  #pragma unroll
  for (int r = 0; r < 8; ++r) {
    float inv = 1.0f / lr[r];
    int m = m0 + ((lane >> 4) << 3) + r;
    int c = ch0 + (lane & 15);
    yTb[(size_t)m * Cc + c]      = (bf16)(O0[r] * inv);
    yTb[(size_t)m * Cc + c + 16] = (bf16)(O1[r] * inv);
  }
}

// ---------------------------------------------------------------------------
// Host launcher
// ---------------------------------------------------------------------------
extern "C" void kernel_launch(void* const* d_in, const int* in_sizes, int n_in,
                              void* d_out, int out_size, void* d_ws, size_t ws_size,
                              hipStream_t stream) {
  (void)in_sizes; (void)n_in; (void)out_size; (void)ws_size;
  const float* x     = (const float*)d_in[0];
  const float* gamma = (const float*)d_in[1];
  const float* beta  = (const float*)d_in[2];
  const float* rmean = (const float*)d_in[3];
  const float* rvar  = (const float*)d_in[4];
  const float* vw    = (const float*)d_in[5];
  const float* zw    = (const float*)d_in[6];
  const float* qw    = (const float*)d_in[7];
  const float* ow    = (const float*)d_in[8];

  // workspace carve-up (bf16 elements), all regions 16B-aligned
  bf16* base = (bf16*)d_ws;
  const size_t n1 = (size_t)Bn * Ms * Cc;        // 1,179,648
  bf16* xnT = base;                              // [b][m][c]
  bf16* vT  = xnT + n1;                          // [b][m][c]
  bf16* zT  = vT  + n1;                          // [b][m][c]
  bf16* qcm = zT  + n1;                          // [b][c][m]
  bf16* zrT = qcm + n1;                          // [b][m][c]
  bf16* yT  = zrT + n1;                          // [b][m][c]
  bf16* rl2 = yT  + n1;                          // [m][c]
  bf16* wb  = rl2 + (size_t)Ms * Cc;             // 4 x 256x256
  bf16* wv = wb;
  bf16* wz = wb + (size_t)Cc * Cc;
  bf16* wq = wb + (size_t)2 * Cc * Cc;
  bf16* wo = wb + (size_t)3 * Cc * Cc;

  cvtw_kernel<<<dim3(4 * Cc * Cc / 256), 256, 0, stream>>>(vw, zw, qw, ow, wb);
  bn_kernel  <<<dim3(Bn * Ms * Cc / 256), 256, 0, stream>>>(x, gamma, beta, rmean, rvar, xnT);
  pe_kernel  <<<dim3(Ms * Cc / 256), 256, 0, stream>>>(rl2);

  // v, z, q projections (WMMA GEMMs)
  gemm_kernel<<<dim3(TILES / 4, 1, Bn), 128, 0, stream>>>(wv, xnT, nullptr, vT, nullptr);
  gemm_kernel<<<dim3(TILES / 4, 1, Bn), 128, 0, stream>>>(wz, xnT, nullptr, zT, nullptr);
  gemm_kernel<<<dim3(TILES / 4, 1, Bn), 128, 0, stream>>>(wq, xnT, qcm, nullptr, nullptr);

  norm_kernel<<<dim3(Bn * Ms / 256), 256, 0, stream>>>(zT, rl2, zrT);

  // flash attention: 144 row-tiles x 8 heads x 2 batches, 4 waves/block
  attn_kernel<<<dim3(MT / 4, HEADS, Bn), 128, 0, stream>>>(vT, zrT, qcm, yT);

  // final o-projection straight into d_out (fp32, channel-major == BCHW)
  gemm_kernel<<<dim3(TILES / 4, 1, Bn), 128, 0, stream>>>(wo, yT, nullptr, nullptr,
                                                          (float*)d_out);
}
